// SSIMLoss_8607114461716
// MI455X (gfx1250) — matrix operand order repaired and checked by reading, not compiled, not run on hardware
//
#include <hip/hip_runtime.h>
#include <hip/hip_bf16.h>

typedef __attribute__((ext_vector_type(16))) _Float16 v16h;
typedef __attribute__((ext_vector_type(8)))  _Float16 v8h;
typedef __attribute__((ext_vector_type(8)))  float    v8f;

#define IMG_W 512
#define IMG_H 512
#define N_PIX 12582912.0   // 16*3*512*512
#define HB_STRIDE 56       // halves per column in transposed H-blur buffer

// ---------------------------------------------------------------------------
// init: zero the double accumulator in workspace
// ---------------------------------------------------------------------------
__global__ void ssim_init_kernel(double* __restrict__ accum) {
    if (threadIdx.x == 0) accum[0] = 0.0;
}

// ---------------------------------------------------------------------------
// main fused SSIM kernel: one 32x32 output tile per 128-thread block.
// Separable 11-tap Gaussian blur done as banded-matrix WMMA (f16 in, f32 acc).
// ---------------------------------------------------------------------------
__global__ __launch_bounds__(128) void ssim_tile_kernel(
    const float* __restrict__ pred,
    const float* __restrict__ targ,
    double* __restrict__ accum)
{
    // LDS staging: 5 quantities {p, t, p*p, t*t, p*t}
    __shared__ __align__(16) _Float16 sIn[5][48 * 48];          // 23040 B
    // horizontally blurred, TRANSPOSED [col][row], stride 56 halves (112 B)
    __shared__ __align__(16) _Float16 sHbT[5][32 * HB_STRIDE];  // 17920 B
    __shared__ float sRed[4];

    const int tid  = threadIdx.x;
    const int lane = tid & 31;
    const int wave = tid >> 5;
    const int h    = lane >> 4;   // lane half (0/1)
    const int n    = lane & 15;   // position within half

    const int c0 = blockIdx.x * 32;
    const int r0 = blockIdx.y * 32;
    const long plane = (long)blockIdx.z * (IMG_W * IMG_H);
    const float* __restrict__ pp = pred + plane;
    const float* __restrict__ tp = targ + plane;

    // Gaussian normalization 1/sum(exp(-(i-5)^2 / (2*1.5^2)))
    float inv = 0.0f;
#pragma unroll
    for (int i = 0; i < 11; ++i)
        inv += __expf(-(float)((i - 5) * (i - 5)) / 4.5f);
    inv = 1.0f / inv;

    // ---------------- load 48x48 halo window (zero padded) ----------------
    for (int idx = tid; idx < 48 * 48; idx += 128) {
        const int lr = idx / 48;
        const int lc = idx - lr * 48;
        const int gr = r0 - 5 + lr;
        const int gc = c0 - 5 + lc;
        float p = 0.0f, t = 0.0f;
        if (gr >= 0 && gr < IMG_H && gc >= 0 && gc < IMG_W) {
            p = pp[gr * IMG_W + gc];
            t = tp[gr * IMG_W + gc];
        }
        sIn[0][idx] = (_Float16)p;
        sIn[1][idx] = (_Float16)t;
        sIn[2][idx] = (_Float16)(p * p);
        sIn[3][idx] = (_Float16)(t * t);
        sIn[4][idx] = (_Float16)(p * t);
    }
    __syncthreads();

    // ------- stage-1 constant band matrix as B operand: B[k][j]=g[k-j] -----
    // B 32x16 f16 layout: halves 0..7 -> K=8h+i, 8..15 -> K=16+8h+(i-8), N = lane&15
    v16h Bband{};
#pragma unroll
    for (int i = 0; i < 16; ++i) {
        const int k = (i < 8) ? (8 * h + i) : (16 + 8 * h + (i - 8));
        const int d = k - n;
        float v = 0.0f;
        if (d >= 0 && d <= 10)
            v = __expf(-(float)((d - 5) * (d - 5)) / 4.5f) * inv;
        Bband[i] = (_Float16)v;
    }

    // ---------------- stage 1: horizontal blur via WMMA --------------------
    // Hout[r, cb+j] = sum_k In[r, cb+k] * g[k-j]   (band in B)
    // A 16x32 f16 layout: m = lane&15; halves 0..7 -> K=8h+i, 8..15 -> K=16+8h+i
    for (int pos = wave; pos < 6; pos += 4) {
        const int rt = pos >> 1;       // row tile 0..2
        const int ct = pos & 1;        // col tile 0..1
        const int abase = (rt * 16 + n) * 48 + ct * 16 + 8 * h;

        v16h A[5];
#pragma unroll
        for (int q = 0; q < 5; ++q) {
            const v8h lo = *(const v8h*)&sIn[q][abase];
            const v8h hi = *(const v8h*)&sIn[q][abase + 16];
#pragma unroll
            for (int i = 0; i < 8; ++i) { A[q][i] = lo[i]; A[q][8 + i] = hi[i]; }
        }

        v8f D[5];
#pragma unroll
        for (int q = 0; q < 5; ++q) {
            v8f Z{};
            D[q] = __builtin_amdgcn_wmma_f32_16x16x32_f16(
                       false, A[q], false, Bband, (short)0, Z, false, false);
        }

        // D layout: VGPR r -> M = r + 8h, N = lane&15.
        // Transposed store: col = ct*16+n, rows rt*16+8h .. +7 contiguous.
        const int sbase = (ct * 16 + n) * HB_STRIDE + rt * 16 + 8 * h;
#pragma unroll
        for (int q = 0; q < 5; ++q) {
            v8h o;
#pragma unroll
            for (int r = 0; r < 8; ++r) o[r] = (_Float16)D[q][r];
            *(v8h*)&sHbT[q][sbase] = o;
        }
    }
    __syncthreads();

    // ------- stage-2 constant band matrix as A operand: A[i][k]=g[k-i] -----
    v16h Aband{};
#pragma unroll
    for (int i = 0; i < 16; ++i) {
        const int k = 16 * h + i;
        const int d = k - n;           // m == lane&15 == n
        float v = 0.0f;
        if (d >= 0 && d <= 10)
            v = __expf(-(float)((d - 5) * (d - 5)) / 4.5f) * inv;
        Aband[i] = (_Float16)v;
    }

    // ---------------- stage 2: vertical blur via WMMA ----------------------
    // Out[rb+i, cb+j] = sum_k g[k-i] * Hout[rb+k, cb+j]
    // B 32x16: half i -> K = 16h+i -> row rb+16h+i, col cb+n (contiguous in sHbT)
    const int rb = (wave >> 1) * 16;
    const int cb = (wave & 1) * 16;
    const int bbase = (cb + n) * HB_STRIDE + rb + 16 * h;

    v16h B[5];
#pragma unroll
    for (int q = 0; q < 5; ++q) {
        const v8h lo = *(const v8h*)&sHbT[q][bbase];
        const v8h hi = *(const v8h*)&sHbT[q][bbase + 8];
#pragma unroll
        for (int i = 0; i < 8; ++i) { B[q][i] = lo[i]; B[q][8 + i] = hi[i]; }
    }

    v8f Dq[5];
#pragma unroll
    for (int q = 0; q < 5; ++q) {
        v8f Z{};
        Dq[q] = __builtin_amdgcn_wmma_f32_16x16x32_f16(
                    false, Aband, false, B[q], (short)0, Z, false, false);
    }

    // ---------------- SSIM map + reduction ---------------------------------
    const float C1 = 1e-4f;   // 0.01^2
    const float C2 = 9e-4f;   // 0.03^2
    float acc = 0.0f;
#pragma unroll
    for (int r = 0; r < 8; ++r) {
        const float mu1 = Dq[0][r], mu2 = Dq[1][r];
        const float e11 = Dq[2][r], e22 = Dq[3][r], e12 = Dq[4][r];
        const float mu1s = mu1 * mu1, mu2s = mu2 * mu2, mu12 = mu1 * mu2;
        const float s1  = e11 - mu1s;
        const float s2  = e22 - mu2s;
        const float s12 = e12 - mu12;
        const float num = (2.0f * mu12 + C1) * (2.0f * s12 + C2);
        const float den = (mu1s + mu2s + C1) * (s1 + s2 + C2);
        acc += num / den;
    }

    // in-wave butterfly reduction (wave32)
#pragma unroll
    for (int off = 16; off > 0; off >>= 1)
        acc += __shfl_xor(acc, off, 32);
    if (lane == 0) sRed[wave] = acc;
    __syncthreads();
    if (tid == 0)
        atomicAdd(accum, (double)(sRed[0] + sRed[1] + sRed[2] + sRed[3]));
}

// ---------------------------------------------------------------------------
// finalize: 1 - mean(ssim_map)
// ---------------------------------------------------------------------------
__global__ void ssim_final_kernel(const double* __restrict__ accum,
                                  float* __restrict__ out) {
    if (threadIdx.x == 0)
        out[0] = 1.0f - (float)(accum[0] / N_PIX);
}

// ---------------------------------------------------------------------------
extern "C" void kernel_launch(void* const* d_in, const int* in_sizes, int n_in,
                              void* d_out, int out_size, void* d_ws, size_t ws_size,
                              hipStream_t stream) {
    const float* pred = (const float*)d_in[0];
    const float* targ = (const float*)d_in[1];
    float* out    = (float*)d_out;
    double* accum = (double*)d_ws;   // 8 bytes used

    ssim_init_kernel<<<1, 32, 0, stream>>>(accum);

    dim3 grid(IMG_W / 32, IMG_H / 32, 48);   // 16 x 16 tiles x (16*3) planes
    ssim_tile_kernel<<<grid, 128, 0, stream>>>(pred, targ, accum);

    ssim_final_kernel<<<1, 32, 0, stream>>>(accum, out);
}
